// GroupViTTokenAssign_12610023981543
// MI455X (gfx1250) — compile-verified, matrix operand order checked by reading, not run on hardware
//
#include <hip/hip_runtime.h>

typedef __attribute__((ext_vector_type(16))) __bf16 v16bf;
typedef __attribute__((ext_vector_type(8)))  __bf16 v8bf;
typedef __attribute__((ext_vector_type(8)))  float  v8f;

#define USE_ASYNC_STAGING 1

namespace cfg {
constexpr int B = 256, L = 196, G = 64, GOUT = 8, D = 384, H = 6;
constexpr int TOK = 192, CH = 1536, HD = 64;
constexpr int GPAD = 32;   // mixer fc2 output padded 8 -> 32 so every GEMM has N%32==0
}

static __device__ __forceinline__ float gelu_exact(float x) {
  return 0.5f * x * (1.0f + erff(x * 0.7071067811865475f));
}

// ---------------------------------------------------------------------------
// WMMA GEMM: C[M,N] = act(A[M,K] @ Wt^T + bias) * scale (+ addsrc)
// Wt is the weight pre-transposed+converted to bf16, layout [N,K] row-major.
// Requirements (guaranteed by launcher): M%128==0, N%32==0, K%64==0.
// Block = 256 threads = 8 waves; each wave owns a 16(M) x 32(N) output strip.
// K advances 64 per barrier pair. W slab (32n x 64k bf16 = 4KB) is staged in
// LDS by one 16B async global->LDS copy per thread (no VGPR round-trip, no
// conversion: rows of Wt are already contiguous bf16 k-runs). Each lane pulls
// B fragments as 16B ds_load_b128; A fragments are pipelined global b128
// loads + packed f32->bf16 converts. 4 v_wmma_f32_16x16x32_bf16 / iteration.
// ---------------------------------------------------------------------------
template<int ACT>   // 0 = none, 1 = exact GELU
__global__ __launch_bounds__(256)
void gemm_wmma_bf16(const float* __restrict__ A, const __bf16* __restrict__ Wt,
                    const float* __restrict__ bias, const float* __restrict__ addsrc,
                    float* __restrict__ C, int M, int N, int K, float scale)
{
  __shared__ __bf16 lds_w[32][64];   // [n_local][k_local], 4 KB

  const int lane  = threadIdx.x & 31;
  const int wave  = threadIdx.x >> 5;
  const int tileM = (blockIdx.y * 8 + wave) * 16;
  const int nbase = blockIdx.x * 32;

  const int idx16 = lane & 15;
  const int khalf = (lane >> 4) * 8;          // 0 or 8
  const float* arow = A + (size_t)(tileM + idx16) * K;

  // staging ownership: one 16B (8 bf16 along k) chunk per thread
  const int n_l  = threadIdx.x >> 3;          // 0..31
  const int koff = (threadIdx.x & 7) * 8;     // 0,8,...,56
  const __bf16* wrow = Wt + (size_t)(nbase + n_l) * K + koff;
#if USE_ASYNC_STAGING
  const unsigned lds_off = (unsigned)(uintptr_t)&lds_w[n_l][koff];  // addr[31:0] = LDS offset
#endif

  v8f acc0 = {};
  v8f acc1 = {};

  const __bf16* w0 = &lds_w[idx16][0];
  const __bf16* w1 = &lds_w[16 + idx16][0];

  for (int k0 = 0; k0 < K; k0 += 64) {
    __syncthreads();  // previous iteration's LDS reads complete
#if USE_ASYNC_STAGING
    {
      const unsigned long long ga = (unsigned long long)(uintptr_t)(wrow + k0);
      asm volatile("global_load_async_to_lds_b128 %0, %1, off"
                   :: "v"(lds_off), "v"(ga) : "memory");
      asm volatile("s_wait_asynccnt 0x0" ::: "memory");
    }
#else
    *(uint4*)&lds_w[n_l][koff] = *(const uint4*)(wrow + k0);
#endif
    __syncthreads();

    if (k0 + 64 < K)
      __builtin_prefetch(arow + k0 + 64 + khalf, 0, 1);   // global_prefetch_b8

#pragma unroll
    for (int kk = 0; kk < 64; kk += 32) {
      // A fragment: contiguous runs k0+kk+khalf+0..7 and +16..23
      const float4* ap = (const float4*)(arow + k0 + kk + khalf);
      const float4 a0 = ap[0], a1 = ap[1], a2 = ap[4], a3 = ap[5];
      v16bf a;
      a[0]  = (__bf16)a0.x; a[1]  = (__bf16)a0.y; a[2]  = (__bf16)a0.z; a[3]  = (__bf16)a0.w;
      a[4]  = (__bf16)a1.x; a[5]  = (__bf16)a1.y; a[6]  = (__bf16)a1.z; a[7]  = (__bf16)a1.w;
      a[8]  = (__bf16)a2.x; a[9]  = (__bf16)a2.y; a[10] = (__bf16)a2.z; a[11] = (__bf16)a2.w;
      a[12] = (__bf16)a3.x; a[13] = (__bf16)a3.y; a[14] = (__bf16)a3.z; a[15] = (__bf16)a3.w;

      // B fragments from LDS: row n_local, two contiguous 8-bf16 (16B) chunks
      const v8bf b0lo = *(const v8bf*)(w0 + kk + khalf);
      const v8bf b0hi = *(const v8bf*)(w0 + kk + khalf + 16);
      const v8bf b1lo = *(const v8bf*)(w1 + kk + khalf);
      const v8bf b1hi = *(const v8bf*)(w1 + kk + khalf + 16);
      const v16bf b0 = __builtin_shufflevector(b0lo, b0hi, 0,1,2,3,4,5,6,7,8,9,10,11,12,13,14,15);
      const v16bf b1 = __builtin_shufflevector(b1lo, b1hi, 0,1,2,3,4,5,6,7,8,9,10,11,12,13,14,15);

      acc0 = __builtin_amdgcn_wmma_f32_16x16x32_bf16(false, a, false, b0, (short)0, acc0, false, false);
      acc1 = __builtin_amdgcn_wmma_f32_16x16x32_bf16(false, a, false, b1, (short)0, acc1, false, false);
    }
  }

  // epilogue: C/D layout — N = lane&15, M = r + (lane>=16 ? 8 : 0)
  const int mbase = tileM + (lane >> 4) * 8;
#pragma unroll
  for (int t = 0; t < 2; ++t) {
    const int n = nbase + t * 16 + idx16;
    const float bv = bias ? bias[n] : 0.0f;
    const v8f& acc = t ? acc1 : acc0;
#pragma unroll
    for (int r = 0; r < 8; ++r) {
      const int m = mbase + r;
      float val = acc[r] + bv;
      if (ACT == 1) val = gelu_exact(val);
      val *= scale;
      if (addsrc) val += addsrc[(size_t)m * N + n];
      C[(size_t)m * N + n] = val;
    }
  }
}

// one-time: transpose+convert weight [K,N] f32 -> [N,K] bf16
__global__ __launch_bounds__(256)
void wt_convert(const float* __restrict__ src, __bf16* __restrict__ dst, int K, int N)
{
  const int i = blockIdx.x * 256 + threadIdx.x;
  if (i < N * K) {
    const int n = i / K, k = i % K;
    dst[i] = (__bf16)src[(size_t)k * N + n];
  }
}

// one-time: mixer fc2 [TOK,GOUT] f32 -> padded transposed [GPAD,TOK] bf16; bias pad
__global__ __launch_bounds__(256)
void pad_fc2_T(const float* __restrict__ w, const float* __restrict__ b,
               __bf16* __restrict__ wt, float* __restrict__ bp)
{
  using namespace cfg;
  const int i = blockIdx.x * 256 + threadIdx.x;
  if (i < GPAD * TOK) {
    const int n = i / TOK, k = i % TOK;
    wt[i] = (__bf16)((n < GOUT) ? w[k * GOUT + n] : 0.0f);
  }
  if (i < GPAD) bp[i] = (i < GOUT) ? b[i] : 0.0f;
}

// ---------------------------------------------------------------------------
// LayerNorm kernels (row length n, biased variance, eps = 1e-5)
// ---------------------------------------------------------------------------
__device__ __forceinline__ void ln_reduce(float a1, float a2, float* s1, float* s2,
                                          float& mean, float& inv, int n)
{
  const int t = threadIdx.x;
  s1[t] = a1; s2[t] = a2;
  __syncthreads();
  for (int off = 64; off > 0; off >>= 1) {
    if (t < off) { s1[t] += s1[t + off]; s2[t] += s2[t + off]; }
    __syncthreads();
  }
  mean = s1[0] / n;
  const float var = s2[0] / n - mean * mean;
  inv = rsqrtf(var + 1e-5f);
}

__global__ __launch_bounds__(128)
void ln_rows(const float* __restrict__ src, float* __restrict__ dst,
             const float* __restrict__ g, const float* __restrict__ b, int n)
{
  const size_t row = blockIdx.x;
  const float* x = src + row * n;
  float*       y = dst + row * n;
  __shared__ float s1[128], s2[128];
  float a1 = 0.f, a2 = 0.f;
  for (int i = threadIdx.x; i < n; i += 128) { float v = x[i]; a1 += v; a2 += v * v; }
  float mean, inv;
  ln_reduce(a1, a2, s1, s2, mean, inv, n);
  for (int i = threadIdx.x; i < n; i += 128)
    y[i] = (x[i] - mean) * inv * g[i] + b[i];
}

// LN over D of group_tokens[B,G,D], written transposed: dst[(b*D+d)*G + g_i]
__global__ __launch_bounds__(128)
void ln_group_T(const float* __restrict__ src, float* __restrict__ dst,
                const float* __restrict__ g, const float* __restrict__ b)
{
  using namespace cfg;
  const int bb = blockIdx.x / G, gi = blockIdx.x % G;
  const float* x = src + ((size_t)bb * G + gi) * D;
  __shared__ float s1[128], s2[128];
  float a1 = 0.f, a2 = 0.f;
  for (int i = threadIdx.x; i < D; i += 128) { float v = x[i]; a1 += v; a2 += v * v; }
  float mean, inv;
  ln_reduce(a1, a2, s1, s2, mean, inv, D);
  for (int i = threadIdx.x; i < D; i += 128)
    dst[((size_t)bb * D + i) * G + gi] = (x[i] - mean) * inv * g[i] + b[i];
}

// mixer output [B*D, GPAD] -> LN over D per (b, gout) -> dst[(b*GOUT+go)*D + d]
__global__ __launch_bounds__(128)
void ln_mix_T(const float* __restrict__ src, float* __restrict__ dst,
              const float* __restrict__ g, const float* __restrict__ b)
{
  using namespace cfg;
  const int bb = blockIdx.x / GOUT, go = blockIdx.x % GOUT;
  __shared__ float s1[128], s2[128];
  float a1 = 0.f, a2 = 0.f;
  for (int i = threadIdx.x; i < D; i += 128) {
    float v = src[((size_t)bb * D + i) * GPAD + go];
    a1 += v; a2 += v * v;
  }
  float mean, inv;
  ln_reduce(a1, a2, s1, s2, mean, inv, D);
  for (int i = threadIdx.x; i < D; i += 128) {
    float v = src[((size_t)bb * D + i) * GPAD + go];
    dst[((size_t)bb * GOUT + go) * D + i] = (v - mean) * inv * g[i] + b[i];
  }
}

// ---------------------------------------------------------------------------
// Multi-head cross attention core: one block per (b,h). Q pre-scaled by HD^-0.5.
// ---------------------------------------------------------------------------
__global__ __launch_bounds__(256)
void cross_attn_core(const float* __restrict__ q, const float* __restrict__ k,
                     const float* __restrict__ v, float* __restrict__ o)
{
  using namespace cfg;
  const int b = blockIdx.x / H, h = blockIdx.x % H;
  __shared__ float qs[GOUT][HD];
  __shared__ float vs[L][HD];
  __shared__ float sc[GOUT][L];
  const int tid = threadIdx.x;

  for (int i = tid; i < GOUT * HD; i += 256) {
    const int qi = i / HD, d = i % HD;
    qs[qi][d] = q[((size_t)b * GOUT + qi) * D + h * HD + d];
  }
  for (int i = tid; i < L * HD; i += 256) {
    const int s = i / HD, d = i % HD;
    vs[s][d] = v[((size_t)b * L + s) * D + h * HD + d];
  }
  __syncthreads();

  for (int s = tid; s < L; s += 256) {
    float acc[GOUT];
#pragma unroll
    for (int qi = 0; qi < GOUT; ++qi) acc[qi] = 0.f;
    const float* kr = k + ((size_t)b * L + s) * D + h * HD;
    for (int d = 0; d < HD; ++d) {
      const float kv = kr[d];
#pragma unroll
      for (int qi = 0; qi < GOUT; ++qi) acc[qi] += qs[qi][d] * kv;
    }
#pragma unroll
    for (int qi = 0; qi < GOUT; ++qi) sc[qi][s] = acc[qi];
  }
  __syncthreads();

  // softmax: one wave per query row (8 waves == GOUT rows)
  const int wrow = tid >> 5, lane = tid & 31;
  {
    float m = -3.4e38f;
    for (int s = lane; s < L; s += 32) m = fmaxf(m, sc[wrow][s]);
    for (int off = 16; off > 0; off >>= 1) m = fmaxf(m, __shfl_xor(m, off));
    float sum = 0.f;
    for (int s = lane; s < L; s += 32) { float e = expf(sc[wrow][s] - m); sc[wrow][s] = e; sum += e; }
    for (int off = 16; off > 0; off >>= 1) sum += __shfl_xor(sum, off);
    const float invs = 1.0f / sum;
    for (int s = lane; s < L; s += 32) sc[wrow][s] *= invs;
  }
  __syncthreads();

  for (int i = tid; i < GOUT * HD; i += 256) {
    const int qi = i / HD, d = i % HD;
    float acc = 0.f;
    for (int s = 0; s < L; ++s) acc += sc[qi][s] * vs[s][d];
    o[((size_t)b * GOUT + qi) * D + h * HD + d] = acc;
  }
}

// ---------------------------------------------------------------------------
// Assign attention: one block per b. Hard one-hot over groups, eps renorm.
// ---------------------------------------------------------------------------
__global__ __launch_bounds__(256)
void assign_attn_core(const float* __restrict__ q2, const float* __restrict__ k2,
                      const float* __restrict__ v2,
                      float* __restrict__ soft_out,   // [B,GOUT,L]
                      float* __restrict__ out)        // [B,GOUT,D]
{
  using namespace cfg;
  const int b = blockIdx.x;
  __shared__ float qs[GOUT][D];
  __shared__ int   idxl[L];
  __shared__ int   cnt[GOUT];
  const int tid = threadIdx.x;

  if (tid < GOUT) cnt[tid] = 0;
  for (int i = tid; i < GOUT * D; i += 256) {
    const int g = i / D, d = i % D;
    qs[g][d] = q2[((size_t)b * GOUT + g) * D + d];
  }
  __syncthreads();

  const float scale = rsqrtf((float)D);
  for (int l = tid; l < L; l += 256) {
    float acc[GOUT];
#pragma unroll
    for (int g = 0; g < GOUT; ++g) acc[g] = 0.f;
    const float* kr = k2 + ((size_t)b * L + l) * D;
    for (int d = 0; d < D; ++d) {
      const float kv = kr[d];
#pragma unroll
      for (int g = 0; g < GOUT; ++g) acc[g] += qs[g][d] * kv;
    }
    float m = -3.4e38f;
#pragma unroll
    for (int g = 0; g < GOUT; ++g) { acc[g] *= scale; m = fmaxf(m, acc[g]); }
    float sum = 0.f;
#pragma unroll
    for (int g = 0; g < GOUT; ++g) { acc[g] = expf(acc[g] - m); sum += acc[g]; }
    const float inv = 1.0f / sum;
    int amax = 0; float best = -1.0f;
#pragma unroll
    for (int g = 0; g < GOUT; ++g) {
      const float sv = acc[g] * inv;
      soft_out[((size_t)b * GOUT + g) * L + l] = sv;
      if (sv > best) { best = sv; amax = g; }
    }
    idxl[l] = amax;
    atomicAdd(&cnt[amax], 1);
  }
  __syncthreads();

  for (int i = tid; i < GOUT * D; i += 256) {
    const int g = i / D, d = i % D;
    float acc = 0.f;
    for (int l = 0; l < L; ++l)
      if (idxl[l] == g) acc += v2[((size_t)b * L + l) * D + d];
    out[((size_t)b * GOUT + g) * D + d] = acc / ((float)cnt[g] + 1.0f);
  }
}

// ---------------------------------------------------------------------------
// Host launcher
// ---------------------------------------------------------------------------
extern "C" void kernel_launch(void* const* d_in, const int* in_sizes, int n_in,
                              void* d_out, int out_size, void* d_ws, size_t ws_size,
                              hipStream_t stream)
{
  using namespace cfg;
  (void)in_sizes; (void)n_in; (void)out_size; (void)ws_size;

  const float* image_tokens = (const float*)d_in[0];
  const float* group_tokens = (const float*)d_in[1];
  // params flattened in jax pytree order (dict keys sorted):
  // assign{k,proj,q,v}{b,w}=2..9 | mlp_channels{fc1,fc2}{b,w}=10..13 |
  // mlp_inter{fc1,fc2}{b,w}=14..17 | norm_new_x{b,g}=18,19 |
  // norm_post_tokens{b,g}=20,21 | norm_tokens{b,g}=22,23 | norm_x{b,g}=24,25 |
  // pre_assign.attn{k,o,q,v}{b,w}=26..33 | pre_assign.mlp{fc1,fc2}{b,w}=34..37 |
  // pre_assign.norm2{b,g}=38,39 | pre_assign.norm_post{b,g}=40,41
  auto P = [&](int i) { return (const float*)d_in[i]; };

  float* out_newx = (float*)d_out;                       // [B,GOUT,D]
  float* out_attn = out_newx + (size_t)B * GOUT * D;     // [B,GOUT,L]

  // workspace bump allocator (bytes, 256B aligned chunks)
  char* base = (char*)d_ws;
  size_t off = 0;
  auto allocf = [&](size_t n) { float*  p = (float*)(base + off);  off = (off + n * 4 + 255) & ~(size_t)255; return p; };
  auto allocb = [&](size_t n) { __bf16* p = (__bf16*)(base + off); off = (off + n * 2 + 255) & ~(size_t)255; return p; };

  const size_t BLD = (size_t)B * L * D;
  const size_t SML = (size_t)B * GOUT * D;
  float* xln   = allocf(BLD);
  float* kbuf  = allocf(BLD);                  // also hosts gT [B*D, G]
  float* vbuf  = allocf(BLD);                  // also hosts mix1 [B*D, TOK]
  float* mix2  = allocf((size_t)B * D * GPAD);
  float* proj  = allocf(SML);
  float* qbuf  = allocf(SML);
  float* aout  = allocf(SML);
  float* ybuf  = allocf(SML);
  float* tmp   = allocf(SML);
  float* y2    = allocf(SML);
  float* proj2 = allocf(SML);
  float* q2    = allocf(SML);
  float* asgn  = allocf(SML);
  float* nx1   = allocf(SML);
  float* hbuf  = allocf((size_t)B * GOUT * CH);
  float* bpad  = allocf((size_t)GPAD);
  // bf16 [N,K] transposed weights
  const size_t WDD = (size_t)D * D, WDC = (size_t)D * CH;
  __bf16* wt_mix1 = allocb((size_t)TOK * G);
  __bf16* wt_mix2 = allocb((size_t)GPAD * TOK);
  __bf16* wt_pq = allocb(WDD); __bf16* wt_pk = allocb(WDD);
  __bf16* wt_pv = allocb(WDD); __bf16* wt_po = allocb(WDD);
  __bf16* wt_pm1 = allocb(WDC); __bf16* wt_pm2 = allocb(WDC);
  __bf16* wt_aq = allocb(WDD); __bf16* wt_ak = allocb(WDD);
  __bf16* wt_av = allocb(WDD); __bf16* wt_ap = allocb(WDD);
  __bf16* wt_c1 = allocb(WDC); __bf16* wt_c2 = allocb(WDC);
  float* gT   = kbuf;
  float* mix1 = vbuf;

  auto wconv = [&](const float* w, __bf16* wt, int K, int N) {
    wt_convert<<<(N * K + 255) / 256, 256, 0, stream>>>(w, wt, K, N);
  };
  auto gemm = [&](const float* A, const __bf16* Wt, const float* bias, const float* add,
                  float* C, int M, int N, int K, float scale, bool gelu) {
    dim3 grid(N / 32, M / 128);
    if (gelu) gemm_wmma_bf16<1><<<grid, 256, 0, stream>>>(A, Wt, bias, add, C, M, N, K, scale);
    else      gemm_wmma_bf16<0><<<grid, 256, 0, stream>>>(A, Wt, bias, add, C, M, N, K, scale);
  };

  const int Mx = B * L;      // 50176 (392*128)
  const int Mg = B * GOUT;   // 2048
  const int Md = B * D;      // 98304

  // 0) one-time weight transpose+convert to bf16 [N,K]
  wconv(P(15), wt_mix1, G, TOK);
  pad_fc2_T<<<(GPAD * TOK + 255) / 256, 256, 0, stream>>>(P(17), P(16), wt_mix2, bpad);
  wconv(P(31), wt_pq, D, D);  wconv(P(27), wt_pk, D, D);
  wconv(P(33), wt_pv, D, D);  wconv(P(29), wt_po, D, D);
  wconv(P(35), wt_pm1, D, CH); wconv(P(37), wt_pm2, CH, D);
  wconv(P(7),  wt_aq, D, D);  wconv(P(3),  wt_ak, D, D);
  wconv(P(9),  wt_av, D, D);  wconv(P(5),  wt_ap, D, D);
  wconv(P(11), wt_c1, D, CH); wconv(P(13), wt_c2, CH, D);

  // 1) g = LN(group_tokens) transposed; mixer MLP over token dim
  ln_group_T<<<B * G, 128, 0, stream>>>(group_tokens, gT, P(23), P(22));
  gemm(gT,   wt_mix1, P(14), nullptr, mix1, Md, TOK,  G,   1.0f, true);
  gemm(mix1, wt_mix2, bpad,  nullptr, mix2, Md, GPAD, TOK, 1.0f, false);
  ln_mix_T<<<B * GOUT, 128, 0, stream>>>(mix2, proj, P(21), P(20));

  // 2) x = LN(image_tokens)
  ln_rows<<<B * L, 128, 0, stream>>>(image_tokens, xln, P(25), P(24), D);

  // 3) pre-assign cross attention
  gemm(proj, wt_pq, P(30), nullptr, qbuf, Mg, D, D, 0.125f, false);  // Q * HD^-0.5
  gemm(xln,  wt_pk, P(26), nullptr, kbuf, Mx, D, D, 1.0f,   false);  // K
  gemm(xln,  wt_pv, P(32), nullptr, vbuf, Mx, D, D, 1.0f,   false);  // V
  cross_attn_core<<<B * H, 256, 0, stream>>>(qbuf, kbuf, vbuf, aout);
  gemm(aout, wt_po, P(28), proj, ybuf, Mg, D, D, 1.0f, false);       // O + residual

  // 4) pre-assign MLP
  ln_rows<<<Mg, 128, 0, stream>>>(ybuf, tmp, P(39), P(38), D);
  gemm(tmp,  wt_pm1, P(34), nullptr, hbuf, Mg, CH, D,  1.0f, true);
  gemm(hbuf, wt_pm2, P(36), ybuf,    y2,   Mg, D,  CH, 1.0f, false);
  ln_rows<<<Mg, 128, 0, stream>>>(y2, proj2, P(41), P(40), D);

  // 5) assign attention (hard softmax over group dim)
  gemm(proj2, wt_aq, P(6), nullptr, q2,   Mg, D, D, 1.0f, false);
  gemm(xln,   wt_ak, P(2), nullptr, kbuf, Mx, D, D, 1.0f, false);
  gemm(xln,   wt_av, P(8), nullptr, vbuf, Mx, D, D, 1.0f, false);
  assign_attn_core<<<B, 256, 0, stream>>>(q2, kbuf, vbuf, out_attn, asgn);
  gemm(asgn, wt_ap, P(4), proj2, nx1, Mg, D, D, 1.0f, false);

  // 6) channel MLP -> new_x (written straight to d_out)
  ln_rows<<<Mg, 128, 0, stream>>>(nx1, tmp, P(19), P(18), D);
  gemm(tmp,  wt_c1, P(10), nullptr, hbuf,     Mg, CH, D,  1.0f, true);
  gemm(hbuf, wt_c2, P(12), nx1,     out_newx, Mg, D,  CH, 1.0f, false);
}
